// GRUCell_20117626814952
// MI455X (gfx1250) — compile-verified, matrix operand order checked by reading, not run on hardware
//
#include <hip/hip_runtime.h>
#include <hip/hip_bf16.h>

#define B_ 8192
#define D_ 1024
#define H_ 1024
#define O_ 1024

typedef __attribute__((ext_vector_type(16))) __bf16 bf16x16;
typedef __attribute__((ext_vector_type(8)))  float  f32x8;
typedef __attribute__((ext_vector_type(8)))  unsigned int u32x8;
typedef __attribute__((ext_vector_type(4)))  unsigned int u32x4;

// ---------- helpers ----------
__device__ __forceinline__ unsigned short f2bf(float f) {
    unsigned int u = __builtin_bit_cast(unsigned int, f);
    unsigned int r = u + 0x7FFFu + ((u >> 16) & 1u);   // round-to-nearest-even
    return (unsigned short)(r >> 16);
}

__device__ __forceinline__ bf16x16 load_frag(const unsigned short* p) {
    u32x8 u = *(const u32x8*)p;                        // 32B
    return __builtin_bit_cast(bf16x16, u);
}

__device__ __forceinline__ float sigmoidf_(float x) {
    return 1.0f / (1.0f + __expf(-x));
}

#define WMMA_BF16(A, Bm, C) \
    __builtin_amdgcn_wmma_f32_16x16x32_bf16(false, (A), false, (Bm), (short)0, (C), false, false)

// ---------- stage 1a: fp32 -> bf16 elementwise ----------
__global__ __launch_bounds__(256) void convert_bf16(const float* __restrict__ in,
                                                    unsigned short* __restrict__ out,
                                                    int n) {
    int i = blockIdx.x * 256 + threadIdx.x;
    if (i < n) out[i] = f2bf(in[i]);
}

// ---------- stage 1b: fp32 [K,N] -> bf16 [N,K] (LDS tile transpose) ----------
__global__ __launch_bounds__(256) void transpose_bf16(const float* __restrict__ in,
                                                      unsigned short* __restrict__ out,
                                                      int K, int N) {
    __shared__ unsigned short tile[32][33];
    const int n0 = blockIdx.x * 32, k0 = blockIdx.y * 32;
    const int tx = threadIdx.x & 31, ty = threadIdx.x >> 5;   // ty in 0..7
#pragma unroll
    for (int i = 0; i < 4; ++i) {
        int k = k0 + ty + i * 8;
        tile[ty + i * 8][tx] = f2bf(in[(size_t)k * N + n0 + tx]);
    }
    __syncthreads();
#pragma unroll
    for (int i = 0; i < 4; ++i) {
        int n = n0 + ty + i * 8;
        out[(size_t)n * K + k0 + tx] = tile[tx][ty + i * 8];
    }
}

// ---------- stage 2: fused GRU gates GEMM + epilogue ----------
// Block = 256 thr = 8 waves. Block tile 128(M) x 32(N); each wave owns 16x32.
// Weight slab (6 matrices x 32 rows x 32 K = 12KB/step) double-buffered in LDS,
// shared by all 8 waves. 12 WMMAs / wave / K-step.
__global__ __launch_bounds__(256) void gru_gates_wmma(
    const unsigned short* __restrict__ xbf,   // [B,D] bf16
    const unsigned short* __restrict__ hbf,   // [B,H] bf16
    const unsigned short* __restrict__ Wcat,  // [6,H,D] bf16: ir,iz,in,hr,hz,hn (transposed)
    const float* __restrict__ bir, const float* __restrict__ biz,
    const float* __restrict__ bin_, const float* __restrict__ bhn,
    const float* __restrict__ h32,            // [B,H] fp32 (original h)
    float* __restrict__ newh)                 // [B,H] fp32
{
    __shared__ __attribute__((aligned(32))) unsigned short ldsB[2][6 * 32 * 32];

    const int tid  = threadIdx.x;
    const int lane = tid & 31;
    const int wave = tid >> 5;
    const int row0 = blockIdx.x * 128 + wave * 16;   // M (per wave)
    const int col0 = blockIdx.y * 32;                // N (per block)
    const int half = lane >> 4;
    const int l16  = lane & 15;

    const unsigned short* xr = xbf + (size_t)(row0 + l16) * D_ + half * 16;
    const unsigned short* hr = hbf + (size_t)(row0 + l16) * H_ + half * 16;

    // cooperative weight staging: 768 chunks of 16B, 3 per thread
    const int c0 = tid, c1 = tid + 256, c2 = tid + 512;
    auto gaddr = [&](int c, int k) {
        const int g = c >> 7, r = (c >> 2) & 31, q = c & 3;
        return (const u32x4*)(Wcat + (size_t)(g * H_ + col0 + r) * D_ + k + q * 8);
    };

    f32x8 accR0 = {}, accR1 = {}, accZ0 = {}, accZ1 = {};
    f32x8 accNX0 = {}, accNX1 = {}, accNH0 = {}, accNH1 = {};

    u32x4 s0 = *gaddr(c0, 0), s1 = *gaddr(c1, 0), s2 = *gaddr(c2, 0);
    int buf = 0;
    const int fb = l16 * 32 + half * 16;             // frag base (elements) within a 16x32 tile row-slab

    for (int k = 0; k < D_; k += 32) {
        u32x4* dst = (u32x4*)&ldsB[buf][0];          // LDS byte offset == chunk*16 by construction
        dst[c0] = s0; dst[c1] = s1; dst[c2] = s2;
        __syncthreads();
        if (k + 32 < D_) {                           // prefetch next K-slab into registers
            s0 = *gaddr(c0, k + 32);
            s1 = *gaddr(c1, k + 32);
            s2 = *gaddr(c2, k + 32);
        }
        __builtin_prefetch(xr + k + 512, 0, 3);
        __builtin_prefetch(hr + k + 512, 0, 3);

        bf16x16 ax = load_frag(xr + k);
        bf16x16 ah = load_frag(hr + k);
        const unsigned short* lb = &ldsB[buf][0];

        bf16x16 b;
        b = load_frag(lb + 0 * 1024 + fb);        accR0  = WMMA_BF16(ax, b, accR0);
        b = load_frag(lb + 0 * 1024 + 512 + fb);  accR1  = WMMA_BF16(ax, b, accR1);
        b = load_frag(lb + 1 * 1024 + fb);        accZ0  = WMMA_BF16(ax, b, accZ0);
        b = load_frag(lb + 1 * 1024 + 512 + fb);  accZ1  = WMMA_BF16(ax, b, accZ1);
        b = load_frag(lb + 2 * 1024 + fb);        accNX0 = WMMA_BF16(ax, b, accNX0);
        b = load_frag(lb + 2 * 1024 + 512 + fb);  accNX1 = WMMA_BF16(ax, b, accNX1);
        b = load_frag(lb + 3 * 1024 + fb);        accR0  = WMMA_BF16(ah, b, accR0);
        b = load_frag(lb + 3 * 1024 + 512 + fb);  accR1  = WMMA_BF16(ah, b, accR1);
        b = load_frag(lb + 4 * 1024 + fb);        accZ0  = WMMA_BF16(ah, b, accZ0);
        b = load_frag(lb + 4 * 1024 + 512 + fb);  accZ1  = WMMA_BF16(ah, b, accZ1);
        b = load_frag(lb + 5 * 1024 + fb);        accNH0 = WMMA_BF16(ah, b, accNH0);
        b = load_frag(lb + 5 * 1024 + 512 + fb);  accNH1 = WMMA_BF16(ah, b, accNH1);

        buf ^= 1;
    }

    // epilogue; C/D layout: VGPR i -> M = half*8 + i, N = lane&15 (per 16-wide tile)
    const int n0v = col0 + l16;
    const int n1v = n0v + 16;
    const float bir0 = bir[n0v], bir1 = bir[n1v];
    const float biz0 = biz[n0v], biz1 = biz[n1v];
    const float bin0 = bin_[n0v], bin1 = bin_[n1v];
    const float bhn0 = bhn[n0v], bhn1 = bhn[n1v];
#pragma unroll
    for (int i = 0; i < 8; ++i) {
        const int m = row0 + half * 8 + i;
        {
            float r  = sigmoidf_(accR0[i] + bir0);
            float z  = sigmoidf_(accZ0[i] + biz0);
            float nt = tanhf(accNX0[i] + bin0 + r * (accNH0[i] + bhn0));
            float hv = h32[(size_t)m * H_ + n0v];
            newh[(size_t)m * H_ + n0v] = (1.0f - z) * nt + z * hv;
        }
        {
            float r  = sigmoidf_(accR1[i] + bir1);
            float z  = sigmoidf_(accZ1[i] + biz1);
            float nt = tanhf(accNX1[i] + bin1 + r * (accNH1[i] + bhn1));
            float hv = h32[(size_t)m * H_ + n1v];
            newh[(size_t)m * H_ + n1v] = (1.0f - z) * nt + z * hv;
        }
    }
}

// ---------- stage 3: LayerNorm row-wise -> bf16 ----------
__global__ __launch_bounds__(256) void layernorm_bf16(const float* __restrict__ x,
                                                      const float* __restrict__ scale,
                                                      const float* __restrict__ bias,
                                                      unsigned short* __restrict__ out) {
    const int row = blockIdx.x;
    const float* xr = x + (size_t)row * H_;
    float v[4], s = 0.f, ss = 0.f;
#pragma unroll
    for (int i = 0; i < 4; ++i) {
        v[i] = xr[threadIdx.x + i * 256];
        s  += v[i];
        ss += v[i] * v[i];
    }
#pragma unroll
    for (int off = 16; off > 0; off >>= 1) {      // wave32 shuffle reduce
        s  += __shfl_xor(s,  off, 32);
        ss += __shfl_xor(ss, off, 32);
    }
    __shared__ float red0[8], red1[8];
    const int wave = threadIdx.x >> 5;
    if ((threadIdx.x & 31) == 0) { red0[wave] = s; red1[wave] = ss; }
    __syncthreads();
    float ts = 0.f, tss = 0.f;
#pragma unroll
    for (int i = 0; i < 8; ++i) { ts += red0[i]; tss += red1[i]; }
    const float mu  = ts * (1.0f / H_);
    const float var = tss * (1.0f / H_) - mu * mu;
    const float rs  = rsqrtf(var + 1e-6f);
#pragma unroll
    for (int i = 0; i < 4; ++i) {
        int c = threadIdx.x + i * 256;
        out[(size_t)row * H_ + c] = f2bf((v[i] - mu) * rs * scale[c] + bias[c]);
    }
}

// ---------- stage 4: readout GEMM (norm @ Wout + bout) ----------
// Block tile 128(M) x 32(N), wave tile 16x32, Wout slab double-buffered in LDS.
__global__ __launch_bounds__(256) void out_gemm_wmma(
    const unsigned short* __restrict__ abf,    // [B,H] bf16 (normalized)
    const unsigned short* __restrict__ WoutT,  // [O,H] bf16 transposed
    const float* __restrict__ bout,
    float* __restrict__ out)                   // [B,O] fp32
{
    __shared__ __attribute__((aligned(32))) unsigned short ldsB[2][32 * 32];

    const int tid  = threadIdx.x;
    const int lane = tid & 31;
    const int wave = tid >> 5;
    const int row0 = blockIdx.x * 128 + wave * 16;
    const int col0 = blockIdx.y * 32;
    const int half = lane >> 4;
    const int l16  = lane & 15;

    const unsigned short* ar = abf + (size_t)(row0 + l16) * H_ + half * 16;

    // staging: 32 rows x 64B = 128 chunks of 16B; threads 0..127 carry one each
    const int r = (tid >> 2) & 31, q = tid & 3;
    const unsigned short* wsrc = WoutT + (size_t)(col0 + r) * H_ + q * 8;

    f32x8 acc0 = {}, acc1 = {};
    u32x4 s0 = {};
    if (tid < 128) s0 = *(const u32x4*)wsrc;
    int buf = 0;
    const int fb = l16 * 32 + half * 16;

    for (int k = 0; k < H_; k += 32) {
        if (tid < 128) ((u32x4*)&ldsB[buf][0])[tid] = s0;
        __syncthreads();
        if (tid < 128 && k + 32 < H_) s0 = *(const u32x4*)(wsrc + k + 32);
        __builtin_prefetch(ar + k + 512, 0, 3);

        bf16x16 a  = load_frag(ar + k);
        const unsigned short* lb = &ldsB[buf][0];
        bf16x16 b0 = load_frag(lb + fb);
        bf16x16 b1 = load_frag(lb + 512 + fb);
        acc0 = WMMA_BF16(a, b0, acc0);
        acc1 = WMMA_BF16(a, b1, acc1);
        buf ^= 1;
    }

    const int n0v = col0 + l16;
    const int n1v = n0v + 16;
    const float bv0 = bout[n0v], bv1 = bout[n1v];
#pragma unroll
    for (int i = 0; i < 8; ++i) {
        const int m = row0 + half * 8 + i;
        out[(size_t)m * O_ + n0v] = acc0[i] + bv0;
        out[(size_t)m * O_ + n1v] = acc1[i] + bv1;
    }
}

// ---------- host ----------
extern "C" void kernel_launch(void* const* d_in, const int* in_sizes, int n_in,
                              void* d_out, int out_size, void* d_ws, size_t ws_size,
                              hipStream_t stream) {
    const float* x    = (const float*)d_in[0];
    const float* h    = (const float*)d_in[1];
    const float* Wir  = (const float*)d_in[2];
    const float* bir  = (const float*)d_in[3];
    const float* Wiz  = (const float*)d_in[4];
    const float* biz  = (const float*)d_in[5];
    const float* Win  = (const float*)d_in[6];
    const float* bin_ = (const float*)d_in[7];
    const float* Whr  = (const float*)d_in[8];
    const float* Whz  = (const float*)d_in[9];
    const float* Whn  = (const float*)d_in[10];
    const float* bhn  = (const float*)d_in[11];
    const float* lns  = (const float*)d_in[12];
    const float* lnb  = (const float*)d_in[13];
    const float* Wout = (const float*)d_in[14];
    const float* bout = (const float*)d_in[15];

    float* newh = (float*)d_out;                       // [B,H]
    float* out2 = (float*)d_out + (size_t)B_ * H_;     // [B,O]

    // workspace layout (bf16 staging), 256B aligned slices
    char* ws = (char*)d_ws;
    size_t off = 0;
    auto take = [&](size_t bytes) { char* p = ws + off; off += (bytes + 255) & ~(size_t)255; return p; };
    unsigned short* xbf   = (unsigned short*)take((size_t)B_ * D_ * 2);
    unsigned short* hbf   = (unsigned short*)take((size_t)B_ * H_ * 2);
    unsigned short* Wcat  = (unsigned short*)take((size_t)6 * H_ * D_ * 2);  // ir,iz,in,hr,hz,hn
    unsigned short* WoutT = (unsigned short*)take((size_t)O_ * H_ * 2);
    unsigned short* nrmbf = (unsigned short*)take((size_t)B_ * H_ * 2);

    // stage 1: precision conversion + weight transposes into packed slab
    convert_bf16<<<(B_ * D_) / 256, 256, 0, stream>>>(x, xbf, B_ * D_);
    convert_bf16<<<(B_ * H_) / 256, 256, 0, stream>>>(h, hbf, B_ * H_);
    dim3 gt(H_ / 32, D_ / 32);
    transpose_bf16<<<gt, 256, 0, stream>>>(Wir, Wcat + (size_t)0 * H_ * D_, D_, H_);
    transpose_bf16<<<gt, 256, 0, stream>>>(Wiz, Wcat + (size_t)1 * H_ * D_, D_, H_);
    transpose_bf16<<<gt, 256, 0, stream>>>(Win, Wcat + (size_t)2 * H_ * D_, D_, H_);
    transpose_bf16<<<gt, 256, 0, stream>>>(Whr, Wcat + (size_t)3 * H_ * D_, H_, H_);
    transpose_bf16<<<gt, 256, 0, stream>>>(Whz, Wcat + (size_t)4 * H_ * D_, H_, H_);
    transpose_bf16<<<gt, 256, 0, stream>>>(Whn, Wcat + (size_t)5 * H_ * D_, H_, H_);
    transpose_bf16<<<dim3(O_ / 32, H_ / 32), 256, 0, stream>>>(Wout, WoutT, H_, O_);

    // stage 2: fused GRU gates (6 GEMMs) + gate math -> new_h
    dim3 g2(B_ / 128, H_ / 32);
    gru_gates_wmma<<<g2, 256, 0, stream>>>(xbf, hbf, Wcat,
                                           bir, biz, bin_, bhn, h, newh);

    // stage 3: LayerNorm -> bf16
    layernorm_bf16<<<B_, 256, 0, stream>>>(newh, lns, lnb, nrmbf);

    // stage 4: readout GEMM
    dim3 g4(B_ / 128, O_ / 32);
    out_gemm_wmma<<<g4, 256, 0, stream>>>(nrmbf, WoutT, bout, out2);
}
